// AttnDBGNNLayer_58067957842554
// MI455X (gfx1250) — matrix operand order repaired and checked by reading, not compiled, not run on hardware
//
#include <hip/hip_runtime.h>

// ---------------------------------------------------------------------------
// Types / WMMA helpers (CDNA5 gfx1250, wave32)
// ---------------------------------------------------------------------------
typedef __attribute__((ext_vector_type(16))) _Float16 v16h;
typedef __attribute__((ext_vector_type(8)))  _Float16 v8h;
typedef __attribute__((ext_vector_type(8)))  float    v8f;

#define NDIM 8192
#define DDIM 128
#define CDIM 6
#define ROWS 768                   // C*D
#define NCD  (8192 * 768)

__device__ __forceinline__ v8f wmma_f16(v16h a, v16h b, v8f c) {
  return __builtin_amdgcn_wmma_f32_16x16x32_f16(false, a, false, b, (short)0, c,
                                                false, false);
}

__device__ __forceinline__ v8f zero8() {
  v8f z;
#pragma unroll
  for (int i = 0; i < 8; ++i) z[i] = 0.f;
  return z;
}

// 16 contiguous halfs -> v16h (two aligned 16B loads)
__device__ __forceinline__ v16h ld16(const _Float16* p) {
  v8h a = *(const v8h*)p;
  v8h b = *(const v8h*)(p + 8);
  v16h r;
#pragma unroll
  for (int i = 0; i < 8; ++i) { r[i] = a[i]; r[i + 8] = b[i]; }
  return r;
}

// A-matrix (16x32 f16) fragment: prow points at this lane's row (m = lane&15),
// h8 = 8*(lane>>4).
__device__ __forceinline__ v16h ld_afrag(const _Float16* prow, int h8) {
  v16h a;
#pragma unroll
  for (int v = 0; v < 8; ++v) {
    const int k0 = 2 * h8 + 2 * v;
    a[2 * v]     = prow[k0];
    a[2 * v + 1] = prow[k0 + 1];
  }
  return a;
}

// ---------------------------------------------------------------------------
// DPP16 butterfly reductions over each 16-lane half (pure VALU, co-issues
// with the XDL/WMMA pipe; avoids ds_bpermute in the flash hot loop).
//   0xB1 = quad_perm(1,0,3,2)   : pair swap
//   0x4E = quad_perm(2,3,0,1)   : quad-half swap
//   0x141 = row_half_mirror     : pairs 4-groups within 8
//   0x140 = row_mirror          : pairs 8-groups within 16
// ---------------------------------------------------------------------------
template <int CTRL>
__device__ __forceinline__ float dppf(float x) {
  const int r = __builtin_amdgcn_update_dpp(
      0, __builtin_bit_cast(int, x), CTRL, 0xF, 0xF, true);
  return __builtin_bit_cast(float, r);
}
__device__ __forceinline__ float redmax16(float x) {
  x = fmaxf(x, dppf<0xB1>(x));
  x = fmaxf(x, dppf<0x4E>(x));
  x = fmaxf(x, dppf<0x141>(x));
  x = fmaxf(x, dppf<0x140>(x));
  return x;
}
__device__ __forceinline__ float redsum16(float x) {
  x += dppf<0xB1>(x);
  x += dppf<0x4E>(x);
  x += dppf<0x141>(x);
  x += dppf<0x140>(x);
  return x;
}

// ---------------------------------------------------------------------------
// Generic tiled GEMM:  out[M,N] = concat_k(A_seg) @ concat_k(W_seg^T) + bias
// fp32 in (A row-stride lda; W row-major [N x 128] per segment), f16 LDS
// tiles, v_wmma_f32_16x16x32_f16, double-buffered staging.
// Block: 256 threads = 8 waves; tile 128(M) x 64(N); wave = 64x16.
// ---------------------------------------------------------------------------
struct GemmArgs {
  const float* A[4];
  const float* W[4];
  int          lda[4];
  const float* bias;
  const float* bias2;
  int          Ktot;
  int          ostride;
  float        qscale;   // applied to cols < 128 when OUT_HALF (Q pre-scale)
};

template <bool OUT_HALF>
__global__ __launch_bounds__(256) void gemm_f16w(GemmArgs g,
                                                 float* __restrict__ outF,
                                                 _Float16* __restrict__ outH) {
  __shared__ _Float16 lA[2][128 * 34];   // [m][k] k-stride 34
  __shared__ _Float16 lBt[2][64 * 40];   // [n][k] k-stride 40 (transposed W)

  const int tid  = threadIdx.x;
  const int wave = tid >> 5;
  const int lane = tid & 31;
  const int c16  = lane & 15;
  const int hh   = lane >> 4;
  const int h8   = hh * 8;
  const int b16  = hh * 16;
  const int wm   = wave & 1;
  const int wn   = wave >> 1;
  const int m0   = blockIdx.x * 128;
  const int n0   = blockIdx.y * 64;

  float ra[16], rb[8];
  auto gload = [&](int ks) {
    const int seg = ks >> 7;
    const int kl  = ks & 127;
    const float* Ab = g.A[seg];
    const float* Wb = g.W[seg];
    const int    lda = g.lda[seg];
#pragma unroll
    for (int ii = 0; ii < 16; ++ii) {
      const int i = tid + ii * 256;
      const int kk = i & 31, mm = i >> 5;
      ra[ii] = Ab[(size_t)(m0 + mm) * lda + kl + kk];
    }
#pragma unroll
    for (int ii = 0; ii < 8; ++ii) {
      const int i = tid + ii * 256;
      const int kk = i & 31, nn = i >> 5;
      rb[ii] = Wb[(size_t)(n0 + nn) * 128 + kl + kk];
    }
  };
  auto lstore = [&](int b) {
#pragma unroll
    for (int ii = 0; ii < 16; ++ii) {
      const int i = tid + ii * 256;
      const int kk = i & 31, mm = i >> 5;
      lA[b][mm * 34 + kk] = (_Float16)ra[ii];
    }
#pragma unroll
    for (int ii = 0; ii < 8; ++ii) {
      const int i = tid + ii * 256;
      const int kk = i & 31, nn = i >> 5;
      lBt[b][nn * 40 + kk] = (_Float16)rb[ii];
    }
  };

  v8f acc[4];
#pragma unroll
  for (int mt = 0; mt < 4; ++mt) acc[mt] = zero8();

  const int nk = g.Ktot >> 5;
  gload(0);
  lstore(0);
  __syncthreads();

  for (int t = 0; t < nk; ++t) {
    if (t + 1 < nk) gload((t + 1) * 32);   // overlap next tile with compute
    const int bb = t & 1;

    const v16h b = ld16(&lBt[bb][(wn * 16 + c16) * 40 + b16]);
#pragma unroll
    for (int mt = 0; mt < 4; ++mt) {
      const v16h a = ld_afrag(&lA[bb][(wm * 64 + mt * 16 + c16) * 34], h8);
      acc[mt] = wmma_f16(a, b, acc[mt]);
    }
    __syncthreads();
    if (t + 1 < nk) {
      lstore((t + 1) & 1);
      __syncthreads();
    }
  }

#pragma unroll
  for (int mt = 0; mt < 4; ++mt) {
#pragma unroll
    for (int r = 0; r < 8; ++r) {
      const int row = m0 + wm * 64 + mt * 16 + r + 8 * hh;
      const int n   = n0 + wn * 16 + c16;
      float v = acc[mt][r];
      if (g.bias)  v += g.bias[n];
      if (g.bias2) v += g.bias2[n];
      if (OUT_HALF) {
        if (n < 128) v *= g.qscale;
        outH[(size_t)row * g.ostride + n] = (_Float16)v;
      } else {
        outF[(size_t)row * g.ostride + n] = v;
      }
    }
  }
}

// ---------------------------------------------------------------------------
// Flash attention (dense softmax over 8192 keys), one 16-row query tile per
// wave, 4 waves per block sharing double-buffered K/V LDS staging.
// qkv is f16 [N x 384] (Q pre-scaled by 1/sqrt(128)); o = softmax(QK^T) V.
// ---------------------------------------------------------------------------
__global__ __launch_bounds__(128) void flash_attn_f16(
    const _Float16* __restrict__ qkv, float* __restrict__ o) {
  __shared__ _Float16 Ks[2][32 * 136];   // [key][d]   d-stride 136
  __shared__ _Float16 Vt[2][128 * 40];   // [d][key]   key-stride 40 (transposed)
  __shared__ _Float16 Ps[4 * 16 * 32];   // per-wave P tile (16 q x 32 keys)

  const int tid  = threadIdx.x;
  const int wave = tid >> 5;
  const int lane = tid & 31;
  const int c16  = lane & 15;
  const int hh   = lane >> 4;
  const int h8   = hh * 8;
  const int b16  = hh * 16;
  const int qt   = blockIdx.x * 4 + wave;

  // Q fragments: 4 K-chunks of 32 over d=128 (loaded once)
  v16h qf[4];
  {
    const _Float16* qp = qkv + (size_t)(qt * 16 + c16) * 384;
#pragma unroll
    for (int cch = 0; cch < 4; ++cch) qf[cch] = ld_afrag(qp + 32 * cch, h8);
  }

  v8f acc[8];
#pragma unroll
  for (int n = 0; n < 8; ++n) acc[n] = zero8();
  float mrow[8], lrow[8];
#pragma unroll
  for (int r = 0; r < 8; ++r) { mrow[r] = -1e30f; lrow[r] = 0.f; }

  _Float16* P    = &Ps[wave * 512];
  const int skey = tid >> 2;          // 0..31 : key this thread stages
  const int sd0  = (tid & 3) * 32;    // d-group of 32

  v8h kreg[4], vreg[4];
  auto gloadKV = [&](int kb) {
    const _Float16* ksrc = qkv + (size_t)(kb + skey) * 384 + 128 + sd0;
#pragma unroll
    for (int i = 0; i < 4; ++i) kreg[i] = *(const v8h*)(ksrc + 8 * i);
    const _Float16* vsrc = qkv + (size_t)(kb + skey) * 384 + 256 + sd0;
#pragma unroll
    for (int i = 0; i < 4; ++i) vreg[i] = *(const v8h*)(vsrc + 8 * i);
  };
  auto lstoreKV = [&](int b) {
#pragma unroll
    for (int i = 0; i < 4; ++i)
      *(v8h*)&Ks[b][skey * 136 + sd0 + 8 * i] = kreg[i];
#pragma unroll
    for (int i = 0; i < 4; ++i)
#pragma unroll
      for (int j = 0; j < 8; ++j)
        Vt[b][(sd0 + 8 * i + j) * 40 + skey] = vreg[i][j];
  };

  const int T = NDIM / 32;
  gloadKV(0);
  lstoreKV(0);
  __syncthreads();

  for (int kt = 0; kt < T; ++kt) {
    if (kt + 1 < T) gloadKV((kt + 1) * 32);   // overlap with compute
    const int bb = kt & 1;

    // S = Q @ K^T : two 16x16 tiles (32 keys)
    v8f s0 = zero8(), s1 = zero8();
#pragma unroll
    for (int cch = 0; cch < 4; ++cch) {
      const v16h bk0 = ld16(&Ks[bb][(0  + c16) * 136 + 32 * cch + b16]);
      const v16h bk1 = ld16(&Ks[bb][(16 + c16) * 136 + 32 * cch + b16]);
      s0 = wmma_f16(qf[cch], bk0, s0);
      s1 = wmma_f16(qf[cch], bk1, s1);
    }

    // online softmax (rows live in 16-lane halves of the C layout); DPP-only
    float alpha[8];
#pragma unroll
    for (int r = 0; r < 8; ++r) {
      const float mx   = redmax16(fmaxf(s0[r], s1[r]));
      const float mnew = fmaxf(mrow[r], mx);
      const float a  = __expf(mrow[r] - mnew);
      const float p0 = __expf(s0[r] - mnew);
      const float p1 = __expf(s1[r] - mnew);
      lrow[r] = lrow[r] * a + redsum16(p0 + p1);
      mrow[r] = mnew;
      alpha[r] = a;
      s0[r] = p0;
      s1[r] = p1;
    }
#pragma unroll
    for (int n = 0; n < 8; ++n)
#pragma unroll
      for (int r = 0; r < 8; ++r) acc[n][r] *= alpha[r];

    // C-layout -> A-layout relayout of P through per-wave LDS (DS in-order)
#pragma unroll
    for (int r = 0; r < 8; ++r) {
      P[(r + 8 * hh) * 32 + c16]      = (_Float16)s0[r];
      P[(r + 8 * hh) * 32 + 16 + c16] = (_Float16)s1[r];
    }
    __builtin_amdgcn_wave_barrier();
    const v16h pf = ld_afrag(&P[c16 * 32], h8);

    // O += P @ V : B fragment from transposed V (key rows contiguous)
#pragma unroll
    for (int n = 0; n < 8; ++n) {
      const v16h bv = ld16(&Vt[bb][(n * 16 + c16) * 40 + b16]);
      acc[n] = wmma_f16(pf, bv, acc[n]);
    }

    __syncthreads();
    if (kt + 1 < T) {
      lstoreKV((kt + 1) & 1);
      __syncthreads();
    }
  }

  float inv[8];
#pragma unroll
  for (int r = 0; r < 8; ++r) inv[r] = 1.0f / lrow[r];
#pragma unroll
  for (int n = 0; n < 8; ++n) {
#pragma unroll
    for (int r = 0; r < 8; ++r) {
      const int row = qt * 16 + r + 8 * hh;
      o[(size_t)row * 128 + n * 16 + c16] = acc[n][r] * inv[r];
    }
  }
}

// ---------------------------------------------------------------------------
// SAGE aggregation: agg[dst] += h[src] per edge (segment-sum via L2 atomics)
// ---------------------------------------------------------------------------
__global__ __launch_bounds__(256) void sage_scatter(
    const float* __restrict__ h, const int* __restrict__ ei,
    float* __restrict__ agg, int E) {
  const int tid = blockIdx.x * 256 + threadIdx.x;
  if (tid >= E * 32) return;
  const int e = tid >> 5, c = tid & 31;
  const int src = ei[e];
  const int dst = ei[E + e];
  const float4 v = *(const float4*)(h + (size_t)src * 128 + (size_t)c * 4);
  float* a = agg + (size_t)dst * 128 + (size_t)c * 4;
  atomicAdd(a + 0, v.x);
  atomicAdd(a + 1, v.y);
  atomicAdd(a + 2, v.z);
  atomicAdd(a + 3, v.w);
}

// ---------------------------------------------------------------------------
// Copy columns 1..5 of x_A / x_B straight into the output tensors
// ---------------------------------------------------------------------------
__global__ __launch_bounds__(256) void copy_rest(const float* __restrict__ xA,
                                                 const float* __restrict__ xB,
                                                 float* __restrict__ out) {
  const int tid = blockIdx.x * 256 + threadIdx.x;
  const int per = NDIM * 160;        // float4 groups per type (5*128/4 per row)
  const int type = tid / per;
  if (type >= 2) return;
  const int rem = tid - type * per;
  const int n = rem / 160, q = rem % 160;
  const float* src = (type ? xB : xA) + (size_t)n * ROWS + 128 + (size_t)q * 4;
  float* dst = out + (size_t)type * NCD + (size_t)n * ROWS + 128 + (size_t)q * 4;
  *(float4*)dst = *(const float4*)src;
}

// ---------------------------------------------------------------------------
// Host launcher
// ---------------------------------------------------------------------------
extern "C" void kernel_launch(void* const* d_in, const int* in_sizes, int n_in,
                              void* d_out, int out_size, void* d_ws,
                              size_t ws_size, hipStream_t stream) {
  (void)n_in; (void)out_size; (void)ws_size;
  const int E = in_sizes[2] / 2;

  const float* xA     = (const float*)d_in[0];
  const float* xB     = (const float*)d_in[1];
  const int*   eiAB   = (const int*)d_in[2];
  const int*   eiBA   = (const int*)d_in[3];
  const int*   eiAA   = (const int*)d_in[4];
  const float* inW_A  = (const float*)d_in[5];
  const float* inB_A  = (const float*)d_in[6];
  const float* outW_A = (const float*)d_in[7];
  const float* outB_A = (const float*)d_in[8];
  const float* inW_B  = (const float*)d_in[9];
  const float* inB_B  = (const float*)d_in[10];
  const float* outW_B = (const float*)d_in[11];
  const float* outB_B = (const float*)d_in[12];
  const float* wl_AB  = (const float*)d_in[13];
  const float* bl_AB  = (const float*)d_in[14];
  const float* wr_AB  = (const float*)d_in[15];
  const float* wl_BA  = (const float*)d_in[16];
  const float* bl_BA  = (const float*)d_in[17];
  const float* wr_BA  = (const float*)d_in[18];
  const float* wl_AA  = (const float*)d_in[19];
  const float* bl_AA  = (const float*)d_in[20];
  const float* wr_AA  = (const float*)d_in[21];
  float* out = (float*)d_out;

  char* ws = (char*)d_ws;
  _Float16* qkvA = (_Float16*)ws;           ws += (size_t)NDIM * 384 * 2;
  _Float16* qkvB = (_Float16*)ws;           ws += (size_t)NDIM * 384 * 2;
  float* oA    = (float*)ws;                ws += (size_t)NDIM * 128 * 4;
  float* oB    = (float*)ws;                ws += (size_t)NDIM * 128 * 4;
  float* hA    = (float*)ws;                ws += (size_t)NDIM * 128 * 4;
  float* hB    = (float*)ws;                ws += (size_t)NDIM * 128 * 4;
  float* aggAB = (float*)ws;                ws += (size_t)NDIM * 128 * 4;
  float* aggBA = (float*)ws;                ws += (size_t)NDIM * 128 * 4;
  float* aggAA = (float*)ws;                ws += (size_t)NDIM * 128 * 4;

  const float qscale = 0.08838834764831845f;  // 1/sqrt(128)

  // 1) QKV projections -> f16 (Q pre-scaled)
  {
    GemmArgs g{}; g.A[0] = xA; g.W[0] = inW_A; g.lda[0] = ROWS;
    g.bias = inB_A; g.bias2 = nullptr; g.Ktot = 128; g.ostride = 384; g.qscale = qscale;
    gemm_f16w<true><<<dim3(64, 6), 256, 0, stream>>>(g, nullptr, qkvA);
  }
  {
    GemmArgs g{}; g.A[0] = xB; g.W[0] = inW_B; g.lda[0] = ROWS;
    g.bias = inB_B; g.bias2 = nullptr; g.Ktot = 128; g.ostride = 384; g.qscale = qscale;
    gemm_f16w<true><<<dim3(64, 6), 256, 0, stream>>>(g, nullptr, qkvB);
  }

  // 2) flash attention
  flash_attn_f16<<<dim3(NDIM / 64), 128, 0, stream>>>(qkvA, oA);
  flash_attn_f16<<<dim3(NDIM / 64), 128, 0, stream>>>(qkvB, oB);

  // 3) output projections -> h (fp32)
  {
    GemmArgs g{}; g.A[0] = oA; g.W[0] = outW_A; g.lda[0] = 128;
    g.bias = outB_A; g.bias2 = nullptr; g.Ktot = 128; g.ostride = 128; g.qscale = 1.f;
    gemm_f16w<false><<<dim3(64, 2), 256, 0, stream>>>(g, hA, nullptr);
  }
  {
    GemmArgs g{}; g.A[0] = oB; g.W[0] = outW_B; g.lda[0] = 128;
    g.bias = outB_B; g.bias2 = nullptr; g.Ktot = 128; g.ostride = 128; g.qscale = 1.f;
    gemm_f16w<false><<<dim3(64, 2), 256, 0, stream>>>(g, hB, nullptr);
  }

  // 4) segment sums
  (void)hipMemsetAsync(aggAB, 0, (size_t)NDIM * 128 * 4, stream);
  (void)hipMemsetAsync(aggBA, 0, (size_t)NDIM * 128 * 4, stream);
  (void)hipMemsetAsync(aggAA, 0, (size_t)NDIM * 128 * 4, stream);
  {
    const int blocks = (E * 32 + 255) / 256;
    sage_scatter<<<blocks, 256, 0, stream>>>(hA, eiAB, aggAB, E);
    sage_scatter<<<blocks, 256, 0, stream>>>(hB, eiBA, aggBA, E);
    sage_scatter<<<blocks, 256, 0, stream>>>(hA, eiAA, aggAA, E);
  }

  // 5) out_B = aggAB @ wl_AB^T + bl_AB + hB @ wr_AB^T  (K = 256 concat)
  {
    GemmArgs g{};
    g.A[0] = aggAB; g.W[0] = wl_AB; g.lda[0] = 128;
    g.A[1] = hB;    g.W[1] = wr_AB; g.lda[1] = 128;
    g.bias = bl_AB; g.bias2 = nullptr; g.Ktot = 256; g.ostride = ROWS; g.qscale = 1.f;
    gemm_f16w<false><<<dim3(64, 2), 256, 0, stream>>>(g, out + (size_t)NCD, nullptr);
  }
  // 6) out_A = aggBA@wl_BA^T + hA@wr_BA^T + aggAA@wl_AA^T + hA@wr_AA^T + biases
  {
    GemmArgs g{};
    g.A[0] = aggBA; g.W[0] = wl_BA; g.lda[0] = 128;
    g.A[1] = hA;    g.W[1] = wr_BA; g.lda[1] = 128;
    g.A[2] = aggAA; g.W[2] = wl_AA; g.lda[2] = 128;
    g.A[3] = hA;    g.W[3] = wr_AA; g.lda[3] = 128;
    g.bias = bl_BA; g.bias2 = bl_AA; g.Ktot = 512; g.ostride = ROWS; g.qscale = 1.f;
    gemm_f16w<false><<<dim3(64, 2), 256, 0, stream>>>(g, out, nullptr);
  }

  // 7) pass-through columns 1..5
  copy_rest<<<(2 * NDIM * 160) / 256, 256, 0, stream>>>(xA, xB, out);
}
